// OrthoLinear_25984552141525
// MI455X (gfx1250) — compile-verified, matrix operand order checked by reading, not run on hardware
//
#include <hip/hip_runtime.h>
#include <hip/hip_bf16.h>

typedef unsigned short ushort_t;
typedef __attribute__((ext_vector_type(16))) __bf16 v16bf;
typedef __attribute__((ext_vector_type(8)))  float  v8f;
typedef int v4i __attribute__((vector_size(4 * sizeof(int))));

#define TOK   8192
#define KDIM  4096
#define NDIM  4096
#define BM    128
#define BN    128
#define BK    32
#define NK    (KDIM / BK)
#define LDSROW 40   /* padded row stride (elems) to spread LDS banks */

#if defined(__has_builtin)
#  if __has_builtin(__builtin_amdgcn_global_load_async_to_lds_b128) && \
      __has_builtin(__builtin_amdgcn_s_wait_asynccnt)
#    define USE_ASYNC 1
#  endif
#endif
#ifndef USE_ASYNC
#  define USE_ASYNC 0
#endif

// ---------------------------------------------------------------------------
// Kernel 1: per-output-row dequant (int4 + scale) fused with CSR residual add.
// One workgroup per row; row accumulated in LDS f32, written out as bf16.
// ---------------------------------------------------------------------------
__global__ __launch_bounds__(256) void dequant_fuse_kernel(
    const int* __restrict__ base_packed,   // [NDIM, KDIM/2] (one byte per int)
    const float* __restrict__ scales,      // [NDIM]
    const float* __restrict__ ortho_vals,  // [nnz]
    const int* __restrict__ ortho_idx,     // [nnz]
    const int* __restrict__ ortho_ptr,     // [NDIM+1]
    __hip_bfloat16* __restrict__ Wb)       // [NDIM, KDIM] bf16 out
{
    __shared__ float row[KDIM];
    const int o = blockIdx.x;
    const float s = scales[o];
    const int half = KDIM / 2;

    for (int c = threadIdx.x; c < half; c += blockDim.x) {
        int p  = base_packed[o * half + c];
        int lo = (p & 0xF) - 8;
        int hi = ((p >> 4) & 0xF) - 8;
        row[2 * c]     = (float)lo * s;
        row[2 * c + 1] = (float)hi * s;
    }
    __syncthreads();

    const int b = ortho_ptr[o], e = ortho_ptr[o + 1];
    for (int k = b + (int)threadIdx.x; k < e; k += blockDim.x) {
        atomicAdd(&row[ortho_idx[k]], ortho_vals[k]);   // ds_add_f32
    }
    __syncthreads();

    for (int c = threadIdx.x; c < KDIM; c += blockDim.x) {
        Wb[(size_t)o * KDIM + c] = __float2bfloat16(row[c]);
    }
}

// ---------------------------------------------------------------------------
// Kernel 2: X f32 -> bf16 (vectorized).
// ---------------------------------------------------------------------------
__global__ __launch_bounds__(256) void convert_x_kernel(
    const float* __restrict__ x, __hip_bfloat16* __restrict__ xb, int n4)
{
    int i = blockIdx.x * blockDim.x + threadIdx.x;
    int stride = gridDim.x * blockDim.x;
    for (; i < n4; i += stride) {
        float4 v = ((const float4*)x)[i];
        union { __hip_bfloat16 h[4]; uint2 u; } pk;
        pk.h[0] = __float2bfloat16(v.x);
        pk.h[1] = __float2bfloat16(v.y);
        pk.h[2] = __float2bfloat16(v.z);
        pk.h[3] = __float2bfloat16(v.w);
        ((uint2*)xb)[i] = pk.u;
    }
}

// ---------------------------------------------------------------------------
// Kernel 3: WMMA GEMM  out[M,N] = Xb[M,K] * Wb[N,K]^T  (bf16 in, f32 acc/out)
// Block tile 128x128, 8 waves (4 down x 2 across), wave tile 32x64 = 2x4 WMMA.
// ---------------------------------------------------------------------------
__global__ __launch_bounds__(256) void gemm_wmma_kernel(
    const ushort_t* __restrict__ Xb,   // [TOK, KDIM] bf16 bits
    const ushort_t* __restrict__ Wb,   // [NDIM, KDIM] bf16 bits
    float* __restrict__ out)           // [TOK, NDIM]
{
    __shared__ __align__(16) ushort_t As[2][BM * LDSROW];
    __shared__ __align__(16) ushort_t Bs[2][BN * LDSROW];

    const int tid  = threadIdx.x;
    const int lane = tid & 31;
    const int wave = tid >> 5;      // 0..7
    const int wm   = wave & 3;      // row group: wm*32
    const int wn   = wave >> 2;     // col group: wn*64

    const int bm = blockIdx.x * BM;
    const int bn = blockIdx.y * BN;

    v8f acc[2][4] = {};

    // ---- staging: 512 x 16B chunks per tile; 2 chunks/thread for A and B ----
#if USE_ASYNC
    auto stage = [&](int kt, int b) {
        const int k0 = kt * BK;
#pragma unroll
        for (int i = 0; i < 2; ++i) {
            int c   = tid + i * 256;
            int r   = c >> 2;
            int seg = (c & 3) * 8;
            const ushort_t* ga = Xb + (size_t)(bm + r) * KDIM + k0 + seg;
            const ushort_t* gb = Wb + (size_t)(bn + r) * KDIM + k0 + seg;
            __builtin_amdgcn_global_load_async_to_lds_b128(
                (__attribute__((address_space(1))) v4i*)ga,
                (__attribute__((address_space(3))) v4i*)&As[b][r * LDSROW + seg],
                0, 0);
            __builtin_amdgcn_global_load_async_to_lds_b128(
                (__attribute__((address_space(1))) v4i*)gb,
                (__attribute__((address_space(3))) v4i*)&Bs[b][r * LDSROW + seg],
                0, 0);
        }
    };
#else
    auto stage = [&](int kt, int b) {
        const int k0 = kt * BK;
#pragma unroll
        for (int i = 0; i < 2; ++i) {
            int c   = tid + i * 256;
            int r   = c >> 2;
            int seg = (c & 3) * 8;
            const ushort_t* ga = Xb + (size_t)(bm + r) * KDIM + k0 + seg;
            const ushort_t* gb = Wb + (size_t)(bn + r) * KDIM + k0 + seg;
            uint4 va = *(const uint4*)ga;
            uint4 vb = *(const uint4*)gb;
            *(uint4*)&As[b][r * LDSROW + seg] = va;
            *(uint4*)&Bs[b][r * LDSROW + seg] = vb;
        }
    };
#endif

    // ---- per-K-step compute from LDS buffer b ----
    auto compute = [&](int b) {
        union Frag { uint4 u[2]; v16bf v; };
        const int lh = lane & 15;
        const int ks = (lane >> 4) << 3;   // 0 or 8, per 16-bit A/B lane layout
        Frag a[2], bb[4];
#pragma unroll
        for (int i = 0; i < 2; ++i) {
            const ushort_t* p = &As[b][(wm * 32 + i * 16 + lh) * LDSROW + ks];
            a[i].u[0] = *(const uint4*)p;          // K = ks..ks+7
            a[i].u[1] = *(const uint4*)(p + 16);   // K = ks+16..ks+23
        }
#pragma unroll
        for (int j = 0; j < 4; ++j) {
            const ushort_t* p = &Bs[b][(wn * 64 + j * 16 + lh) * LDSROW + ks];
            bb[j].u[0] = *(const uint4*)p;
            bb[j].u[1] = *(const uint4*)(p + 16);
        }
#pragma unroll
        for (int i = 0; i < 2; ++i)
#pragma unroll
            for (int j = 0; j < 4; ++j)
                acc[i][j] = __builtin_amdgcn_wmma_f32_16x16x32_bf16(
                    false, a[i].v, false, bb[j].v,
                    (short)0, acc[i][j], false, false);
    };

    // ---- main loop ----
#if USE_ASYNC
    stage(0, 0);
    int buf = 0;
    for (int kt = 0; kt < NK; ++kt) {
        if (kt + 1 < NK) {
            stage(kt + 1, buf ^ 1);
            __builtin_amdgcn_s_wait_asynccnt(4);   // current buffer landed
        } else {
            __builtin_amdgcn_s_wait_asynccnt(0);
        }
        __syncthreads();
        compute(buf);
        __syncthreads();
        buf ^= 1;
    }
#else
    for (int kt = 0; kt < NK; ++kt) {
        __syncthreads();
        stage(kt, 0);
        __syncthreads();
        compute(0);
    }
#endif

    // ---- epilogue: C/D layout -> f32 stores ----
    const int rbase = (lane >> 4) * 8;
    const int col0  = lane & 15;
#pragma unroll
    for (int i = 0; i < 2; ++i)
#pragma unroll
        for (int j = 0; j < 4; ++j) {
            const int row0 = bm + wm * 32 + i * 16 + rbase;
            const int c    = bn + wn * 64 + j * 16 + col0;
#pragma unroll
            for (int r = 0; r < 8; ++r)
                out[(size_t)(row0 + r) * NDIM + c] = acc[i][j][r];
        }
}

// ---------------------------------------------------------------------------
extern "C" void kernel_launch(void* const* d_in, const int* in_sizes, int n_in,
                              void* d_out, int out_size, void* d_ws, size_t ws_size,
                              hipStream_t stream) {
    const float* x           = (const float*)d_in[0];
    const int*   base_packed = (const int*)d_in[1];
    const float* scales      = (const float*)d_in[2];
    const float* ortho_vals  = (const float*)d_in[3];
    const int*   ortho_idx   = (const int*)d_in[4];
    const int*   ortho_ptr   = (const int*)d_in[5];
    float*       out         = (float*)d_out;

    __hip_bfloat16* Wb = (__hip_bfloat16*)d_ws;
    __hip_bfloat16* Xb = (__hip_bfloat16*)((char*)d_ws +
                          (size_t)NDIM * KDIM * sizeof(__hip_bfloat16));

    dequant_fuse_kernel<<<NDIM, 256, 0, stream>>>(
        base_packed, scales, ortho_vals, ortho_idx, ortho_ptr, Wb);

    const int n4 = (TOK * KDIM) / 4;
    convert_x_kernel<<<(n4 + 255) / 256, 256, 0, stream>>>(x, Xb, n4);

    dim3 grid(TOK / BM, NDIM / BN);
    gemm_wmma_kernel<<<grid, 256, 0, stream>>>(
        (const ushort_t*)Xb, (const ushort_t*)Wb, out);
}
